// FusePart_89395449299264
// MI455X (gfx1250) — matrix operand order repaired and checked by reading, not compiled, not run on hardware
//
#include <hip/hip_runtime.h>
#include <hip/hip_bf16.h>
#include <stdint.h>

typedef __attribute__((ext_vector_type(16))) _Float16 v16h;
typedef __attribute__((ext_vector_type(8)))  float    v8f;

// ---------------------------------------------------------------------------
// WMMA fragment loaders (layouts per CDNA5 ISA 7.12.2, wave32)
// A: 16x32 f16, row-major source (lda halves per row), lanes 0-15 hold K {0-7,16-23},
//    lanes 16-31 hold K {8-15,24-31}, each VGPR = one contiguous K-pair -> dword loads.
// B: 32x16 f16 from K-contiguous "Bt" storage Bt[n][k] (ldb halves per N-row),
//    lanes 0-15 K {0-7,16-23}, lanes 16-31 K {8-15,24-31}, N = lane&15.
// ---------------------------------------------------------------------------
__device__ __forceinline__ v16h load_a_frag(const _Float16* A, int lda, int kk, int lane) {
    int row = lane & 15;
    int b0  = (lane >> 4) << 2;   // dword base 0 or 4  (K-half base 0 or 8)
    const uint32_t* p32 = reinterpret_cast<const uint32_t*>(A + (size_t)row * lda + kk);
    union { v16h h; uint32_t u[8]; } fr;
#pragma unroll
    for (int r = 0; r < 4; ++r) fr.u[r] = p32[b0 + r];          // K = kb..kb+7
#pragma unroll
    for (int r = 0; r < 4; ++r) fr.u[4 + r] = p32[8 + b0 + r];  // K = 16+kb..16+kb+7
    return fr.h;
}

__device__ __forceinline__ v16h load_b_frag(const _Float16* Bt, int ldb, int kk, int lane) {
    int col = lane & 15;
    int b0  = (lane >> 4) << 2;   // dword base 0 or 4  (K-half base 0 or 8)
    const uint32_t* p32 = reinterpret_cast<const uint32_t*>(Bt + (size_t)col * ldb + kk);
    union { v16h h; uint32_t u[8]; } fr;
#pragma unroll
    for (int r = 0; r < 4; ++r) fr.u[r] = p32[b0 + r];
#pragma unroll
    for (int r = 0; r < 4; ++r) fr.u[4 + r] = p32[8 + b0 + r];
    return fr.h;
}

#define WMMA_F16(a, b, c) \
    __builtin_amdgcn_wmma_f32_16x16x32_f16(false, (a), false, (b), (short)0, (c), false, false)

// ---------------------------------------------------------------------------
// Weight conversion kernels
// ---------------------------------------------------------------------------
// fc_bin [7][1536][256] f32 -> Bt1 [7][256][1536] f16 (K-contiguous per output col)
__global__ void convert_fcbin(const float* __restrict__ fc, _Float16* __restrict__ Bt) {
    size_t e = (size_t)blockIdx.x * 256 + threadIdx.x;  // total 7*256*1536, exact
    int d = (int)(e % 1536);
    size_t po = e / 1536;
    int o = (int)(po % 256);
    int p = (int)(po / 256);
    Bt[e] = (_Float16)fc[((size_t)p * 1536 + d) * 256 + o];
}

__global__ void cvt_copy_f16(const float* __restrict__ in, _Float16* __restrict__ out, int n) {
    int i = blockIdx.x * 256 + threadIdx.x;
    if (i < n) out[i] = (_Float16)in[i];
}

// ---------------------------------------------------------------------------
// Fused gather + width pooling (mean+max) + pose pooling + leaky_relu
// fuseA layout: [p][m=ns][d], d<1408: sil part (d = c*22 + hh), d>=1408: pose
// ---------------------------------------------------------------------------
__global__ __launch_bounds__(256) void build_fuse(const float* __restrict__ sil,
                                                  const float* __restrict__ pose,
                                                  const int*   __restrict__ min_idx,
                                                  _Float16*    __restrict__ fuseA) {
    int bm = blockIdx.x;           // 0..3359 == p*480 + m
    int p = bm / 480;
    int m = bm % 480;
    int n = m / 30;
    int s = m % 30;
    for (int d = threadIdx.x; d < 1536; d += 256) {
        float val;
        if (d < 1408) {
            int cc = d / 22, hh = d % 22;
            int h0 = min_idx[p * 16 + n] + hh;
            const float* row = sil + ((((size_t)n * 64 + cc) * 30 + s) * 64 + h0) * 22;
            float sum = 0.f, mx = -3.0e38f;
#pragma unroll
            for (int w = 0; w < 22; ++w) {
                float x = row[w];
                sum += x;
                mx = fmaxf(mx, x);
            }
            val = sum * (1.f / 22.f) + mx;
        } else {
            int d2 = d - 1408;
            int cpc = d2 >> 1, t = d2 & 1;
            float x;
            if (p == 0) {
                // AdaptiveAvgPool1d(2) on 5 joints: mean(j0..2), mean(j2..4)
                int j0 = (t == 0) ? 0 : 2;
                const float* pr = pose + (((size_t)n * 64 + cpc) * 30 + s) * 17;
                x = (pr[j0] + pr[j0 + 1] + pr[j0 + 2]) * (1.f / 3.f);
            } else {
                int j = 5 + (p - 1) * 2 + t;
                x = pose[(((size_t)n * 64 + cpc) * 30 + s) * 17 + j];
            }
            val = x > 0.f ? x : 0.01f * x;   // LeakyReLU(0.01)
        }
        fuseA[(size_t)bm * 1536 + d] = (_Float16)val;
    }
}

// ---------------------------------------------------------------------------
// GEMM1: per-part FC. One wave per 32x32 C tile (2x2 WMMA register blocking).
// x0[token = m*7+p][o] = sum_d fuseA[p][m][d] * fc_bin[p][d][o]
// ---------------------------------------------------------------------------
__global__ __launch_bounds__(32) void gemm_fc(const _Float16* __restrict__ fuseA,
                                              const _Float16* __restrict__ Bt1,
                                              float* __restrict__ x0) {
    int ntile = blockIdx.x;  // 0..7   (32-wide N)
    int mtile = blockIdx.y;  // 0..14  (32-tall M)
    int p     = blockIdx.z;  // 0..6
    int lane  = threadIdx.x;
    const _Float16* A0 = fuseA + ((size_t)p * 480 + (size_t)mtile * 32) * 1536;
    const _Float16* A1 = A0 + 16 * 1536;
    const _Float16* B0 = Bt1 + ((size_t)p * 256 + (size_t)ntile * 32) * 1536;
    const _Float16* B1 = B0 + 16 * 1536;
    v8f c00 = {}, c01 = {}, c10 = {}, c11 = {};
    for (int kk = 0; kk < 1536; kk += 32) {
        v16h a0 = load_a_frag(A0, 1536, kk, lane);
        v16h a1 = load_a_frag(A1, 1536, kk, lane);
        v16h b0 = load_b_frag(B0, 1536, kk, lane);
        v16h b1 = load_b_frag(B1, 1536, kk, lane);
        c00 = WMMA_F16(a0, b0, c00);
        c01 = WMMA_F16(a0, b1, c01);
        c10 = WMMA_F16(a1, b0, c10);
        c11 = WMMA_F16(a1, b1, c11);
    }
    int oc = lane & 15;
    int mr = (lane >> 4) << 3;
#pragma unroll
    for (int r = 0; r < 8; ++r) {
        int m0 = mtile * 32 + mr + r;
        int m1 = m0 + 16;
        int o0 = ntile * 32 + oc;
        int o1 = o0 + 16;
        x0[((size_t)m0 * 7 + p) * 256 + o0] = c00[r];
        x0[((size_t)m0 * 7 + p) * 256 + o1] = c01[r];
        x0[((size_t)m1 * 7 + p) * 256 + o0] = c10[r];
        x0[((size_t)m1 * 7 + p) * 256 + o1] = c11[r];
    }
}

// ---------------------------------------------------------------------------
// BatchNorm: two-stage batch stats over 3360 tokens per channel (biased var).
// ---------------------------------------------------------------------------
__global__ __launch_bounds__(256) void bn_stats_partial(const float* __restrict__ x0,
                                                        float* __restrict__ psum,
                                                        float* __restrict__ psq) {
    int b = blockIdx.x;          // 0..20, 160 tokens each
    int o = threadIdx.x;
    float s = 0.f, s2 = 0.f;
    for (int t = b * 160; t < (b + 1) * 160; ++t) {
        float v = x0[(size_t)t * 256 + o];
        s += v;
        s2 += v * v;
    }
    psum[b * 256 + o] = s;
    psq[b * 256 + o]  = s2;
}

__global__ __launch_bounds__(256) void bn_stats_final(const float* __restrict__ psum,
                                                      const float* __restrict__ psq,
                                                      float* __restrict__ stats) {
    int o = threadIdx.x;
    float s = 0.f, s2 = 0.f;
#pragma unroll
    for (int b = 0; b < 21; ++b) {
        s += psum[b * 256 + o];
        s2 += psq[b * 256 + o];
    }
    float mean = s * (1.f / 3360.f);
    float var  = s2 * (1.f / 3360.f) - mean * mean;
    stats[o]       = mean;
    stats[256 + o] = rsqrtf(var + 1e-5f);
}

__global__ __launch_bounds__(256) void bn_apply(const float* __restrict__ x0,
                                                const float* __restrict__ stats,
                                                const float* __restrict__ gamma,
                                                const float* __restrict__ beta,
                                                float* __restrict__ xbnf,
                                                _Float16* __restrict__ xbnh) {
    int t = blockIdx.x;
    int o = threadIdx.x;
    size_t i = (size_t)t * 256 + o;
    float v = (x0[i] - stats[o]) * stats[256 + o] * gamma[o] + beta[o];
    xbnf[i] = v;
    xbnh[i] = (_Float16)v;
}

// ---------------------------------------------------------------------------
// GEMM2: qkv = xbn @ w_qkv.T (M=3360, K=256, N=768). One wave per 32x32 tile.
// ---------------------------------------------------------------------------
__global__ __launch_bounds__(32) void gemm_qkv(const _Float16* __restrict__ A,
                                               const _Float16* __restrict__ Bt,
                                               float* __restrict__ C) {
    int ntile = blockIdx.x;  // 0..23
    int mtile = blockIdx.y;  // 0..104
    int lane  = threadIdx.x;
    const _Float16* A0 = A  + (size_t)mtile * 32 * 256;
    const _Float16* A1 = A0 + 16 * 256;
    const _Float16* B0 = Bt + (size_t)ntile * 32 * 256;
    const _Float16* B1 = B0 + 16 * 256;
    v8f c00 = {}, c01 = {}, c10 = {}, c11 = {};
    for (int kk = 0; kk < 256; kk += 32) {
        v16h a0 = load_a_frag(A0, 256, kk, lane);
        v16h a1 = load_a_frag(A1, 256, kk, lane);
        v16h b0 = load_b_frag(B0, 256, kk, lane);
        v16h b1 = load_b_frag(B1, 256, kk, lane);
        c00 = WMMA_F16(a0, b0, c00);
        c01 = WMMA_F16(a0, b1, c01);
        c10 = WMMA_F16(a1, b0, c10);
        c11 = WMMA_F16(a1, b1, c11);
    }
    int oc = lane & 15;
    int mr = (lane >> 4) << 3;
#pragma unroll
    for (int r = 0; r < 8; ++r) {
        size_t m0 = (size_t)mtile * 32 + mr + r;
        size_t m1 = m0 + 16;
        int o0 = ntile * 32 + oc;
        int o1 = o0 + 16;
        C[m0 * 768 + o0] = c00[r];
        C[m0 * 768 + o1] = c01[r];
        C[m1 * 768 + o0] = c10[r];
        C[m1 * 768 + o1] = c11[r];
    }
}

// ---------------------------------------------------------------------------
// Attention over 7 part tokens; softmax over the HEAD axis (faithful to ref).
// One block per batch element b (480 blocks).
// ---------------------------------------------------------------------------
__global__ __launch_bounds__(256) void attention(const float* __restrict__ qkvb,
                                                 _Float16* __restrict__ aoh) {
    int b = blockIdx.x;
    __shared__ float q[7 * 256];
    __shared__ float k[7 * 256];
    __shared__ float v[7 * 256];
    __shared__ float sc[7 * 7 * 8];   // [n][m][h]

    for (int idx = threadIdx.x; idx < 7 * 768; idx += 256) {
        int nn = idx / 768;
        int i  = idx % 768;
        float val = qkvb[((size_t)(b * 7 + nn)) * 768 + i];
        if (i < 256)      q[nn * 256 + i]         = val;
        else if (i < 512) k[nn * 256 + (i - 256)] = val;
        else              v[nn * 256 + (i - 512)] = val;
    }
    __syncthreads();

    for (int t = threadIdx.x; t < 392; t += 256) {   // 7*7*8 scores
        int nn = t / 56;
        int rem = t % 56;
        int mm = rem / 8;
        int h  = rem % 8;
        float s = 0.f;
#pragma unroll
        for (int d = 0; d < 32; ++d)
            s += q[nn * 256 + h * 32 + d] * k[mm * 256 + h * 32 + d];
        sc[(nn * 7 + mm) * 8 + h] = s * 0.1767766953f;  // 32^-0.5
    }
    __syncthreads();

    if (threadIdx.x < 49) {           // softmax over h for each (n,m)
        int base = threadIdx.x * 8;
        float mx = sc[base];
#pragma unroll
        for (int h = 1; h < 8; ++h) mx = fmaxf(mx, sc[base + h]);
        float e[8];
        float den = 0.f;
#pragma unroll
        for (int h = 0; h < 8; ++h) { e[h] = __expf(sc[base + h] - mx); den += e[h]; }
        float inv = 1.f / den;
#pragma unroll
        for (int h = 0; h < 8; ++h) sc[base + h] = e[h] * inv;
    }
    __syncthreads();

    for (int t = threadIdx.x; t < 1792; t += 256) {   // ao: 7 tokens x 256 ch
        int nn = t / 256;
        int i  = t % 256;
        int h  = i >> 5;
        float s = 0.f;
#pragma unroll
        for (int mm = 0; mm < 7; ++mm)
            s += sc[(nn * 7 + mm) * 8 + h] * v[mm * 256 + i];
        aoh[((size_t)(b * 7 + nn)) * 256 + i] = (_Float16)s;
    }
}

// ---------------------------------------------------------------------------
// GEMM3: out = xbn + ao @ w_proj.T, scattered to [n][out][s][p].
// One wave per 32x32 tile.
// ---------------------------------------------------------------------------
__device__ __forceinline__ void proj_store(float acc, int tkn, int o,
                                           const float* __restrict__ xbnf,
                                           float* __restrict__ out) {
    float y  = acc + xbnf[(size_t)tkn * 256 + o];
    int nb   = tkn / 210;             // token = (nb*30+s)*7 + p
    int rem  = tkn % 210;
    int s_   = rem / 7;
    int p_   = rem % 7;
    out[(((size_t)nb * 256 + o) * 30 + s_) * 7 + p_] = y;
}

__global__ __launch_bounds__(32) void gemm_proj(const _Float16* __restrict__ aoh,
                                                const _Float16* __restrict__ Bt3,
                                                const float* __restrict__ xbnf,
                                                float* __restrict__ out) {
    int ntile = blockIdx.x;  // 0..7
    int mtile = blockIdx.y;  // 0..104
    int lane  = threadIdx.x;
    const _Float16* A0 = aoh + (size_t)mtile * 32 * 256;
    const _Float16* A1 = A0 + 16 * 256;
    const _Float16* B0 = Bt3 + (size_t)ntile * 32 * 256;
    const _Float16* B1 = B0 + 16 * 256;
    v8f c00 = {}, c01 = {}, c10 = {}, c11 = {};
    for (int kk = 0; kk < 256; kk += 32) {
        v16h a0 = load_a_frag(A0, 256, kk, lane);
        v16h a1 = load_a_frag(A1, 256, kk, lane);
        v16h b0 = load_b_frag(B0, 256, kk, lane);
        v16h b1 = load_b_frag(B1, 256, kk, lane);
        c00 = WMMA_F16(a0, b0, c00);
        c01 = WMMA_F16(a0, b1, c01);
        c10 = WMMA_F16(a1, b0, c10);
        c11 = WMMA_F16(a1, b1, c11);
    }
    int oc = lane & 15;
    int mr = (lane >> 4) << 3;
#pragma unroll
    for (int r = 0; r < 8; ++r) {
        int m0 = mtile * 32 + mr + r;
        int m1 = m0 + 16;
        int o0 = ntile * 32 + oc;
        int o1 = o0 + 16;
        proj_store(c00[r], m0, o0, xbnf, out);
        proj_store(c01[r], m0, o1, xbnf, out);
        proj_store(c10[r], m1, o0, xbnf, out);
        proj_store(c11[r], m1, o1, xbnf, out);
    }
}

// ---------------------------------------------------------------------------
extern "C" void kernel_launch(void* const* d_in, const int* in_sizes, int n_in,
                              void* d_out, int out_size, void* d_ws, size_t ws_size,
                              hipStream_t stream) {
    (void)in_sizes; (void)n_in; (void)out_size; (void)ws_size;
    const float* sil     = (const float*)d_in[0];
    const float* pose    = (const float*)d_in[1];
    const float* fc_bin  = (const float*)d_in[2];
    const float* gamma   = (const float*)d_in[3];
    const float* beta    = (const float*)d_in[4];
    const float* w_qkv   = (const float*)d_in[5];
    const float* w_proj  = (const float*)d_in[6];
    const int*   min_idx = (const int*)d_in[7];
    float*       out     = (float*)d_out;

    char* ws = (char*)d_ws;
    size_t off = 0;
    auto alloc = [&](size_t bytes) -> char* {
        char* p = ws + off;
        off = (off + bytes + 255) & ~(size_t)255;
        return p;
    };
    _Float16* Bt1   = (_Float16*)alloc(7ull * 256 * 1536 * 2);  // fc_bin^T f16
    _Float16* Bt2   = (_Float16*)alloc(768ull * 256 * 2);       // w_qkv f16
    _Float16* Bt3   = (_Float16*)alloc(256ull * 256 * 2);       // w_proj f16
    _Float16* fuseA = (_Float16*)alloc(7ull * 480 * 1536 * 2);  // pooled activations f16
    float*    x0    = (float*)   alloc(3360ull * 256 * 4);      // pre-BN FC output
    float*    stats = (float*)   alloc(512 * 4);                // mean | rstd
    float*    psum  = (float*)   alloc(21ull * 256 * 4);        // BN partial sums
    float*    psq   = (float*)   alloc(21ull * 256 * 4);        // BN partial sq-sums
    float*    xbnf  = (float*)   alloc(3360ull * 256 * 4);      // BN output f32
    _Float16* xbnh  = (_Float16*)alloc(3360ull * 256 * 2);      // BN output f16
    float*    qkvb  = (float*)   alloc(3360ull * 768 * 4);      // qkv
    _Float16* aoh   = (_Float16*)alloc(3360ull * 256 * 2);      // attention output f16

    convert_fcbin<<<10752, 256, 0, stream>>>(fc_bin, Bt1);                      // 7*256*1536/256
    cvt_copy_f16<<<(768 * 256 + 255) / 256, 256, 0, stream>>>(w_qkv, Bt2, 768 * 256);
    cvt_copy_f16<<<(256 * 256 + 255) / 256, 256, 0, stream>>>(w_proj, Bt3, 256 * 256);

    build_fuse<<<3360, 256, 0, stream>>>(sil, pose, min_idx, fuseA);

    gemm_fc<<<dim3(8, 15, 7), 32, 0, stream>>>(fuseA, Bt1, x0);

    bn_stats_partial<<<21, 256, 0, stream>>>(x0, psum, psq);
    bn_stats_final<<<1, 256, 0, stream>>>(psum, psq, stats);
    bn_apply<<<3360, 256, 0, stream>>>(x0, stats, gamma, beta, xbnf, xbnh);

    gemm_qkv<<<dim3(24, 105), 32, 0, stream>>>(xbnh, Bt2, qkvb);

    attention<<<480, 256, 0, stream>>>(qkvb, aoh);

    gemm_proj<<<dim3(8, 105), 32, 0, stream>>>(aoh, Bt3, xbnf, out);
}
